// EdgeAwareActor_87797721465079
// MI455X (gfx1250) — matrix operand order repaired and checked by reading, not compiled, not run on hardware
//
#include <hip/hip_runtime.h>

// GATv2 GNN forward for MI455X (gfx1250, wave32, WMMA + TDM).
// GEMMs: fp32 operands pre-packed once into WMMA per-lane bf16 hi/lo layout;
// inner loop = coalesced b128 loads + 3x v_wmma_f32_16x16x32_bf16 (fp32-ish
// accuracy via hi*hi + lo*hi + hi*lo). Edge passes: wave32 per edge, LDS-staged
// We via tensor_load_to_lds (TDM) + s_wait_tensorcnt.

#define IN_F 64
#define H_F 128
#define OUT_F 32
#define ED_F 16
#define HEADS 2
#define NLAYER 3
#define HH (HEADS * H_F) /* 256 */
#define NEG_SLOPE 0.2f

typedef __attribute__((ext_vector_type(16))) __bf16 v16bf;
typedef __attribute__((ext_vector_type(8))) float v8f;
typedef __attribute__((ext_vector_type(4))) unsigned int v4u;
typedef __attribute__((ext_vector_type(8))) unsigned int v8u;

__device__ __forceinline__ __bf16 bf_from_bits(unsigned short s) {
  __bf16 r; __builtin_memcpy(&r, &s, 2); return r;
}
__device__ __forceinline__ unsigned short bf_rne_bits(float f) {
  unsigned u = __float_as_uint(f);
  u += 0x7FFFu + ((u >> 16) & 1u);
  return (unsigned short)(u >> 16);
}
__device__ __forceinline__ void atomicMaxF(float* addr, float val) {
  int* ai = (int*)addr;
  int cur = *ai;
  while (val > __int_as_float(cur)) {
    int assumed = cur;
    cur = atomicCAS(ai, assumed, __float_as_int(val));
    if (cur == assumed) break;
  }
}

// ---------------------------------------------------------------------------
// Pack A[M,K] fp32 -> WMMA 16x32 bf16 tile layout, hi/lo split.
// Per tile (mt,kt): 32 lanes x (16 hi + 16 lo) bf16 = 64 B per lane chunk,
// contiguous across lanes -> GEMM reads are fully coalesced.
// A-layout: lane<16: row=lane, K = kb..kb+15, kb=0;
// lane>=16: kb=16.
// ---------------------------------------------------------------------------
__global__ void pack_a_bf16(const float* __restrict__ A, __bf16* __restrict__ out,
                            int M, int K) {
  const int lane = threadIdx.x & 31;
  const int w = blockIdx.x * 8 + (threadIdx.x >> 5);
  const int KT = K >> 5;
  if (w >= (M >> 4) * KT) return;
  const int mt = w / KT, kt = w - mt * KT;
  const int half = lane >> 4, l15 = lane & 15;
  const int akb = half * 16;
  const float* Ap = A + (size_t)(mt * 16 + l15) * K + kt * 32;
  __bf16* o = out + ((size_t)w * 32 + lane) * 32;
#pragma unroll
  for (int j = 0; j < 8; ++j) {
    float f0 = Ap[akb + j];
    unsigned u0 = __float_as_uint(f0) & 0xFFFF0000u;
    o[j]      = bf_from_bits((unsigned short)(u0 >> 16));
    o[16 + j] = bf_from_bits(bf_rne_bits(f0 - __uint_as_float(u0)));
    float f1 = Ap[8 + akb + j];
    unsigned u1 = __float_as_uint(f1) & 0xFFFF0000u;
    o[8 + j]  = bf_from_bits((unsigned short)(u1 >> 16));
    o[24 + j] = bf_from_bits(bf_rne_bits(f1 - __uint_as_float(u1)));
  }
}

// Pack B[K,N] fp32 -> WMMA 32x16 bf16 tile layout, hi/lo split.
// B-layout: lane holds col n=lane&15; lane<16: K=k0..k0+15, lane>=16: K=k0+16..k0+31.
__global__ void pack_b_bf16(const float* __restrict__ B, __bf16* __restrict__ out,
                            int K, int N) {
  const int lane = threadIdx.x & 31;
  const int w = blockIdx.x * 8 + (threadIdx.x >> 5);
  const int KT = K >> 5;
  if (w >= (N >> 4) * KT) return;
  const int nt = w / KT, kt = w - nt * KT;
  const int half = lane >> 4, l15 = lane & 15;
  const float* Bp = B + (size_t)(kt * 32 + half * 16) * N + nt * 16 + l15;
  __bf16* o = out + ((size_t)w * 32 + lane) * 32;
#pragma unroll
  for (int j = 0; j < 16; ++j) {
    float f = Bp[(size_t)j * N];
    unsigned u = __float_as_uint(f) & 0xFFFF0000u;
    o[j]      = bf_from_bits((unsigned short)(u >> 16));
    o[16 + j] = bf_from_bits(bf_rne_bits(f - __uint_as_float(u)));
  }
}

// ---------------------------------------------------------------------------
// C[M,N] = act(A @ B + bias) from pre-packed bf16 hi/lo operands.
// One wave32 per 16x16 tile; inner loop: 4x 32B coalesced loads + 3 WMMAs.
// ---------------------------------------------------------------------------
__global__ void gemm_packed_wmma(const __bf16* __restrict__ Ap,
                                 const __bf16* __restrict__ Bp,
                                 const float* __restrict__ bias,
                                 float* __restrict__ C,
                                 int M, int K, int N, int act) {
  const int lane = threadIdx.x & 31;
  const int wave = threadIdx.x >> 5;
  const int ntiles = N >> 4;
  const int tile = blockIdx.x * 8 + wave;
  if (tile >= (M >> 4) * ntiles) return;
  const int mt = tile / ntiles;
  const int nt = tile - mt * ntiles;
  const int KT = K >> 5;
  const v16bf* pa = (const v16bf*)Ap + ((size_t)mt * KT * 32 + lane) * 2;
  const v16bf* pb = (const v16bf*)Bp + ((size_t)nt * KT * 32 + lane) * 2;
  v8f acc = {};
  for (int kt = 0; kt < KT; ++kt) {
    v16bf ah = pa[0];
    v16bf al = pa[1];
    v16bf bh = pb[0];
    v16bf bl2 = pb[1];
    pa += 64; pb += 64; // 32 lanes * 2 v16bf per tile
    acc = __builtin_amdgcn_wmma_f32_16x16x32_bf16(false, ah, false, bh,
                                                  (short)0, acc, false, false);
    acc = __builtin_amdgcn_wmma_f32_16x16x32_bf16(false, al, false, bh,
                                                  (short)0, acc, false, false);
    acc = __builtin_amdgcn_wmma_f32_16x16x32_bf16(false, ah, false, bl2,
                                                  (short)0, acc, false, false);
  }
  const int half = lane >> 4;
  const int bcol = nt * 16 + (lane & 15);
  const float bb = bias ? bias[bcol] : 0.f;
#pragma unroll
  for (int v = 0; v < 8; ++v) {
    int row = mt * 16 + v + half * 8;
    float val = acc[v] + bb;
    if (act == 1) val = fmaxf(val, 0.f);
    else if (act == 2) val = 1.f / (1.f + __expf(-val));
    C[(size_t)row * N + bcol] = val;
  }
}

// ---------------------------------------------------------------------------
__global__ void fill_f32(float* __restrict__ p, float v, int n) {
  int idx = blockIdx.x * blockDim.x + threadIdx.x;
  if (idx < n) p[idx] = v;
}

__global__ void deg_easum(const int* __restrict__ ei, const float* __restrict__ eattr,
                          float* __restrict__ deg, float* __restrict__ easum, int E) {
  int idx = blockIdx.x * blockDim.x + threadIdx.x;
  if (idx >= E * ED_F) return;
  int e = idx >> 4, d = idx & 15;
  int t = ei[E + e];
  atomicAdd(&easum[(size_t)t * ED_F + d], eattr[idx]);
  if (d == 0) atomicAdd(&deg[t], 1.f);
}

__global__ void ea_mean_div(float* __restrict__ eam, const float* __restrict__ deg, int Nn) {
  int idx = blockIdx.x * blockDim.x + threadIdx.x;
  if (idx >= Nn * ED_F) return;
  eam[idx] /= fmaxf(deg[idx >> 4], 1.f);
}

// ---------------------------------------------------------------------------
// Pass A: wave32 per edge. We_l (16KB) staged into LDS by the Tensor Data
// Mover: wave 0 builds a 1-D D# (count=1, 4B elements, tile_dim0=4096) in
// SGPRs and issues tensor_load_to_lds (2-group form, <=2D tensor), then
// s_wait_tensorcnt 0 before the block barrier. att staged by thread loads.
// ---------------------------------------------------------------------------
__global__ void edge_logits(const int* __restrict__ ei, const float* __restrict__ eattr,
                            const float* __restrict__ eam,
                            const float* __restrict__ xl, const float* __restrict__ xr,
                            const float* __restrict__ We_l, const float* __restrict__ att_l,
                            float* __restrict__ logits, float* __restrict__ mx,
                            int E, int Etot) {
  __shared__ float sW[ED_F * HH + HH]; // 16*256 We + 256 att = 17408 B
  // att: plain thread loads (1KB)
  for (int i = threadIdx.x; i < HH; i += blockDim.x)
    sW[ED_F * HH + i] = att_l[i];
  // We: TDM async tensor load, issued once by wave 0 (all operands uniform)
  if ((threadIdx.x >> 5) == 0) {
    unsigned ldsoff = (unsigned)(unsigned long long)(&sW[0]); // LDS byte offset
    unsigned long long ga = (unsigned long long)We_l;
    v4u g0;
    g0.x = 1u;                                   // count=1, user mode
    g0.y = ldsoff;                               // lds_addr
    g0.z = (unsigned)(ga & 0xFFFFFFFFu);         // global_addr[31:0]
    g0.w = (unsigned)((ga >> 32) & 0x01FFFFFFu)  // global_addr[56:32]
           | (2u << 30);                         // type=2 (image)
    v8u g1;
    g1.s0 = 2u << 16;          // data_size=2 (4 bytes); mask/flags = 0
    g1.s1 = 4096u << 16;       // tensor_dim0[15:0]=4096 in bits[31:16]
    g1.s2 = 1u << 16;          // tensor_dim0[31:16]=0; tensor_dim1[15:0]=1
    g1.s3 = 4096u << 16;       // tensor_dim1[31:16]=0; tile_dim0=4096
    g1.s4 = 0u;                // tile_dim1=0 (unused), tile_dim2=0
    g1.s5 = 4096u;             // tensor_dim0_stride[31:0]
    g1.s6 = 0u;                // stride hi, tensor_dim1_stride lo
    g1.s7 = 0u;
    asm volatile("tensor_load_to_lds %0, %1" :: "s"(g0), "s"(g1) : "memory");
    __builtin_amdgcn_s_wait_tensorcnt(0);
  }
  __syncthreads();

  const int lane = threadIdx.x & 31;
  const int wave = threadIdx.x >> 5;
  const int e = blockIdx.x * 8 + wave;
  if (e >= Etot) return;
  int s, t;
  const float* ea;
  if (e < E) { s = ei[e]; t = ei[E + e]; ea = eattr + (size_t)e * ED_F; }
  else       { s = e - E; t = s;         ea = eam + (size_t)s * ED_F; }
  float er[ED_F];
#pragma unroll
  for (int k = 0; k < ED_F; ++k) er[k] = ea[k]; // same addr all lanes: broadcast
  const int c0 = lane * 8; // lanes 0-15 -> head0 channels, 16-31 -> head1
  const float* xls = xl + (size_t)s * HH + c0;
  const float* xrt = xr + (size_t)t * HH + c0;
  float p = 0.f;
#pragma unroll
  for (int j = 0; j < 8; ++j) {
    int c = c0 + j;
    float ef = 0.f;
#pragma unroll
    for (int k = 0; k < ED_F; ++k) ef += er[k] * sW[k * HH + c];
    float sv = xls[j] + xrt[j] + ef;
    float m = (sv > 0.f) ? sv : NEG_SLOPE * sv;
    p += m * sW[ED_F * HH + c];
  }
  for (int off = 1; off < 16; off <<= 1) p += __shfl_xor(p, off, 32);
  if ((lane & 15) == 0) {
    int hh = lane >> 4;
    logits[(size_t)e * 2 + hh] = p;
    atomicMaxF(&mx[t * 2 + hh], p);
  }
}

__global__ void edge_softmax_norm(const int* __restrict__ ei, int E, int Etot,
                                  const float* __restrict__ mx,
                                  float* __restrict__ exbuf, float* __restrict__ dn) {
  int idx = blockIdx.x * blockDim.x + threadIdx.x;
  if (idx >= Etot * 2) return;
  int e = idx >> 1, hh = idx & 1;
  int t = (e < E) ? ei[E + e] : (e - E);
  float v = __expf(exbuf[idx] - mx[t * 2 + hh]);
  exbuf[idx] = v;
  atomicAdd(&dn[t * 2 + hh], v);
}

__global__ void edge_aggregate(const int* __restrict__ ei,
                               const float* __restrict__ xl,
                               const float* __restrict__ exbuf,
                               const float* __restrict__ dn,
                               float* __restrict__ agg, int E, int Etot) {
  const int lane = threadIdx.x & 31;
  const int wave = threadIdx.x >> 5;
  const int e = blockIdx.x * 8 + wave;
  if (e >= Etot) return;
  int s, t;
  if (e < E) { s = ei[e]; t = ei[E + e]; } else { s = e - E; t = s; }
  const int c0 = lane * 8;
  const int hh = c0 >> 7;
  const float alpha = exbuf[(size_t)e * 2 + hh] / dn[t * 2 + hh];
  const float* xls = xl + (size_t)s * HH + c0;
  float* ag = agg + (size_t)t * HH + c0;
#pragma unroll
  for (int j = 0; j < 8; ++j) atomicAdd(&ag[j], alpha * xls[j]);
}

__global__ void node_finalize(const float* __restrict__ agg,
                              const float* __restrict__ bconv_l,
                              float* __restrict__ h, int Nn) {
  int idx = blockIdx.x * blockDim.x + threadIdx.x;
  if (idx >= Nn * H_F) return;
  int n = idx >> 7, c = idx & 127;
  float v = 0.5f * (agg[(size_t)n * HH + c] + agg[(size_t)n * HH + H_F + c]) + bconv_l[c];
  h[idx] = fmaxf(v, 0.f);
}

// ---------------------------------------------------------------------------
extern "C" void kernel_launch(void* const* d_in, const int* in_sizes, int n_in,
                              void* d_out, int out_size, void* d_ws, size_t ws_size,
                              hipStream_t stream) {
  (void)n_in; (void)out_size; (void)ws_size;
  const float* x   = (const float*)d_in[0];
  const int*   ei  = (const int*)d_in[1];   // int32 per harness convention
  const float* ea  = (const float*)d_in[2];
  const float* W0  = (const float*)d_in[3];
  const float* b0  = (const float*)d_in[4];
  const float* Wl  = (const float*)d_in[5];
  const float* bl  = (const float*)d_in[6];
  const float* Wr  = (const float*)d_in[7];
  const float* br  = (const float*)d_in[8];
  const float* We  = (const float*)d_in[9];
  const float* att = (const float*)d_in[10];
  const float* bcv = (const float*)d_in[11];
  const float* W1  = (const float*)d_in[12];
  const float* b1  = (const float*)d_in[13];
  const float* W2  = (const float*)d_in[14];
  const float* b2  = (const float*)d_in[15];
  float* out = (float*)d_out;

  const int Nn = in_sizes[0] / IN_F;
  const int E  = in_sizes[1] / 2;
  const int Etot = E + Nn;

  char* base = (char*)d_ws;
  size_t off = 0;
  auto carve = [&](size_t bytes) {
    void* p = base + off;
    off = (off + bytes + 63) & ~(size_t)63;
    return p;
  };
  float* h    = (float*)carve((size_t)Nn * H_F * 4);
  float* xl   = (float*)carve((size_t)Nn * HH * 4);
  float* xr   = (float*)carve((size_t)Nn * HH * 4);
  float* agg  = (float*)carve((size_t)Nn * HH * 4);
  float* exb  = (float*)carve((size_t)Etot * 2 * 4);
  float* mx   = (float*)carve((size_t)Nn * 2 * 4);
  float* dn   = (float*)carve((size_t)Nn * 2 * 4);
  float* deg  = (float*)carve((size_t)Nn * (1 + ED_F) * 4); // deg | easum contiguous
  float* eam  = deg + Nn;
  float* hid  = (float*)carve((size_t)Nn * H_F * 4);
  __bf16* xpack   = (__bf16*)carve((size_t)Nn * IN_F * 2 * 2);
  __bf16* hpack   = (__bf16*)carve((size_t)Nn * H_F * 2 * 2);
  __bf16* hidpack = (__bf16*)carve((size_t)Nn * H_F * 2 * 2);
  __bf16* W0p = (__bf16*)carve((size_t)IN_F * H_F * 2 * 2);
  __bf16* Wlp = (__bf16*)carve((size_t)NLAYER * H_F * HH * 2 * 2);
  __bf16* Wrp = (__bf16*)carve((size_t)NLAYER * H_F * HH * 2 * 2);
  __bf16* W1p = (__bf16*)carve((size_t)H_F * H_F * 2 * 2);
  __bf16* W2p = (__bf16*)carve((size_t)H_F * OUT_F * 2 * 2);

  auto cdiv = [](int a, int b) { return (a + b - 1) / b; };
  const int T = 256;
  const size_t WSLICE = (size_t)H_F * HH * 2; // packed elems per layer weight

  // ---- pack weights into WMMA bf16 hi/lo layout (once per call) ----
  pack_b_bf16<<<cdiv((H_F/16)*(IN_F/32), 8), T, 0, stream>>>(W0, W0p, IN_F, H_F);
  for (int l = 0; l < NLAYER; ++l) {
    pack_b_bf16<<<cdiv((HH/16)*(H_F/32), 8), T, 0, stream>>>(
        Wl + (size_t)l * H_F * HH, Wlp + (size_t)l * WSLICE, H_F, HH);
    pack_b_bf16<<<cdiv((HH/16)*(H_F/32), 8), T, 0, stream>>>(
        Wr + (size_t)l * H_F * HH, Wrp + (size_t)l * WSLICE, H_F, HH);
  }
  pack_b_bf16<<<cdiv((H_F/16)*(H_F/32), 8), T, 0, stream>>>(W1, W1p, H_F, H_F);
  pack_b_bf16<<<cdiv((OUT_F/16)*(H_F/32), 8), T, 0, stream>>>(W2, W2p, H_F, OUT_F);

  // ---- self-loop edge_attr = per-target mean of incoming edge attrs ----
  fill_f32<<<cdiv(Nn * (1 + ED_F), T), T, 0, stream>>>(deg, 0.f, Nn * (1 + ED_F));
  deg_easum<<<cdiv(E * ED_F, T), T, 0, stream>>>(ei, ea, deg, eam, E);
  ea_mean_div<<<cdiv(Nn * ED_F, T), T, 0, stream>>>(eam, deg, Nn);

  // ---- h = relu(x @ W0 + b0) ----
  pack_a_bf16<<<cdiv((Nn/16)*(IN_F/32), 8), T, 0, stream>>>(x, xpack, Nn, IN_F);
  gemm_packed_wmma<<<cdiv((Nn/16)*(H_F/16), 8), T, 0, stream>>>(
      xpack, W0p, b0, h, Nn, IN_F, H_F, 1);

  for (int l = 0; l < NLAYER; ++l) {
    pack_a_bf16<<<cdiv((Nn/16)*(H_F/32), 8), T, 0, stream>>>(h, hpack, Nn, H_F);
    gemm_packed_wmma<<<cdiv((Nn/16)*(HH/16), 8), T, 0, stream>>>(
        hpack, Wlp + (size_t)l * WSLICE, bl + (size_t)l * HH, xl, Nn, H_F, HH, 0);
    gemm_packed_wmma<<<cdiv((Nn/16)*(HH/16), 8), T, 0, stream>>>(
        hpack, Wrp + (size_t)l * WSLICE, br + (size_t)l * HH, xr, Nn, H_F, HH, 0);

    fill_f32<<<cdiv(Nn * HH, T), T, 0, stream>>>(agg, 0.f, Nn * HH);
    fill_f32<<<cdiv(Nn * 2, T), T, 0, stream>>>(mx, -3.0e38f, Nn * 2);
    fill_f32<<<cdiv(Nn * 2, T), T, 0, stream>>>(dn, 0.f, Nn * 2);

    edge_logits<<<cdiv(Etot, 8), T, 0, stream>>>(
        ei, ea, eam, xl, xr, We + (size_t)l * ED_F * HH, att + (size_t)l * HH,
        exb, mx, E, Etot);
    edge_softmax_norm<<<cdiv(Etot * 2, T), T, 0, stream>>>(ei, E, Etot, mx, exb, dn);
    edge_aggregate<<<cdiv(Etot, 8), T, 0, stream>>>(ei, xl, exb, dn, agg, E, Etot);
    node_finalize<<<cdiv(Nn * H_F, T), T, 0, stream>>>(agg, bcv + (size_t)l * H_F, h, Nn);
  }

  // ---- decoder: relu(h@W1+b1), sigmoid(hid@W2+b2) ----
  pack_a_bf16<<<cdiv((Nn/16)*(H_F/32), 8), T, 0, stream>>>(h, hpack, Nn, H_F);
  gemm_packed_wmma<<<cdiv((Nn/16)*(H_F/16), 8), T, 0, stream>>>(
      hpack, W1p, b1, hid, Nn, H_F, H_F, 1);
  pack_a_bf16<<<cdiv((Nn/16)*(H_F/32), 8), T, 0, stream>>>(hid, hidpack, Nn, H_F);
  gemm_packed_wmma<<<cdiv((Nn/16)*(OUT_F/16), 8), T, 0, stream>>>(
      hidpack, W2p, b2, out, Nn, H_F, OUT_F, 2);
}
